// DifferentiableDAG_8306466750962
// MI455X (gfx1250) — matrix operand order repaired and checked by reading, not compiled, not run on hardware
//
#include <hip/hip_runtime.h>
#include <math.h>

#define Bdim   8
#define Tdim   256
#define Hdim   768
#define NHdim  12
#define HDdim  64
#define Mdim   (Bdim*Tdim)      // 2048
#define Ddim   4
#define NOPSd  4
#define KRAW   3076             // 4*H + NOPS
#define KPAD   3104             // padded to multiple of 32
#define NEG_   (-1.0e9f)
#define SCALE_ 0.125f           // 1/sqrt(64)

typedef __attribute__((ext_vector_type(16))) __bf16 v16bf;
typedef __attribute__((ext_vector_type(8)))  float  v8f;
typedef unsigned int v4u __attribute__((ext_vector_type(4)));
typedef int          v8i __attribute__((ext_vector_type(8)));
typedef int          v4i __attribute__((ext_vector_type(4)));

union U16 { unsigned short u[16]; v16bf v; };

__device__ inline unsigned short f2bf(float f) {
  unsigned int x = __float_as_uint(f);
  unsigned int r = (x + 0x7fffu + ((x >> 16) & 1u)) >> 16;   // RNE
  return (unsigned short)r;
}
__device__ inline float bf2f(unsigned short u) {
  return __uint_as_float(((unsigned int)u) << 16);
}

__device__ inline v8f wmma_bf(const U16& a, const U16& b, v8f c) {
  return __builtin_amdgcn_wmma_f32_16x16x32_bf16(false, a.v, false, b.v,
                                                 (short)0, c, false, false);
}

__device__ inline float gumbelf(unsigned seed, unsigned idx) {
  unsigned x = idx * 0x9E3779B9u + seed * 0x85EBCA6Bu + 0x6C62272Eu;
  x ^= x >> 17; x *= 0xED5AD4BBu; x ^= x >> 11; x *= 0xAC4C1B51u;
  x ^= x >> 15; x *= 0x31848BABu; x ^= x >> 14;
  float u = ((float)(x >> 8) + 0.5f) * (1.0f / 16777216.0f);  // (0,1)
  return -__logf(-__logf(u));
}

// ---------------------------------------------------------------------------
// TDM: async DMA of a 2D bf16 tile (tile_cols x tile_rows, row stride in
// elements) from global memory into LDS.  Descriptor per CDNA5 ISA ch.8.
// Rows >= tensor_rows read as zero (OOB guard for edge N-tiles).
// ---------------------------------------------------------------------------
__device__ inline void tdm_load_2d(unsigned lds_addr, const void* gaddr,
                                   unsigned tile_cols, unsigned tile_rows,
                                   unsigned tensor_rows, unsigned row_stride)
{
  const unsigned long long ga = (unsigned long long)(size_t)gaddr;
  v4u g0 = { 1u,                                    // count=1, user descriptor
             lds_addr,                              // LDS byte address
             (unsigned)(ga & 0xffffffffull),        // global_addr[31:0]
             (unsigned)((ga >> 32) & 0x01ffffffull) | (2u << 30) }; // [56:32]+type=2
  v8i g1 = { (int)(1u << 16),                                        // data_size=2B
             (int)((tile_cols & 0xffffu) << 16),                     // tensor_dim0 lo16
             (int)(((tile_cols >> 16) & 0xffffu) |
                   ((tensor_rows & 0xffffu) << 16)),                 // dim0 hi / dim1 lo
             (int)(((tensor_rows >> 16) & 0xffffu) |
                   ((tile_cols & 0xffffu) << 16)),                   // dim1 hi / tile_dim0
             (int)(tile_rows & 0xffffu),                             // tile_dim1 (dim2=0)
             (int)row_stride,                                        // dim0_stride lo32
             0, 0 };
  v4i gz = { 0, 0, 0, 0 };
#if __clang_major__ >= 23
  v8i gz8 = { 0, 0, 0, 0, 0, 0, 0, 0 };
  __builtin_amdgcn_tensor_load_to_lds(g0, g1, gz, gz, gz8, 0);
#else
  __builtin_amdgcn_tensor_load_to_lds(g0, g1, gz, gz, 0);
#endif
}

// Per-wave fragment assembly + 4 WMMAs on a 32x32 output slab.
__device__ inline void frag_mma(const unsigned short* __restrict__ As,
                                const unsigned short* __restrict__ Bs,
                                int wr, int wc, int half, int r, v8f (&acc)[2][2])
{
  U16 af[2], bfr[2];
#pragma unroll
  for (int mi = 0; mi < 2; ++mi) {
    const int arow = wr * 32 + mi * 16 + r;
#pragma unroll
    for (int i = 0; i < 16; ++i) af[mi].u[i] = As[arow * 32 + half * 16 + i];
  }
#pragma unroll
  for (int ci = 0; ci < 2; ++ci) {
    const int brow = wc * 32 + ci * 16 + r;
#pragma unroll
    for (int i = 0; i < 16; ++i) bfr[ci].u[i] = Bs[brow * 32 + half * 16 + i];
  }
#pragma unroll
  for (int mi = 0; mi < 2; ++mi)
#pragma unroll
    for (int ci = 0; ci < 2; ++ci)
      acc[mi][ci] = wmma_bf(af[mi], bfr[ci], acc[mi][ci]);
}

// ---------------------------------------------------------------------------
// bf16 WMMA GEMM.  Cf/Cb = epi(A[M,K](bf16) @ op(B)(bf16) + bias) (+ resid)
//   transB=1: B = W[N,K] row-major (A @ W^T); double-buffered TDM pipeline.
//   transB=0: B = [K,N] row-major; manual (transposing) bf16 staging.
//   act: 0 none, 1 exact gelu, 2 clamp(+-40).  ldc = C/resid row stride.
// Block tile 128x64x32, 8 waves (4x2); each wave a 32x32 slab = 4 WMMA/step.
// ---------------------------------------------------------------------------
__global__ __launch_bounds__(256)
void gemm_k(const unsigned short* __restrict__ A,
            const unsigned short* __restrict__ Bsrc,
            const float* __restrict__ bias, const float* __restrict__ resid,
            float* __restrict__ Cf, unsigned short* __restrict__ Cb,
            int Mr, int Nr, int Kr, int act, int transB, int ldc,
            long long sA, long long sB, long long sC)
{
  __shared__ alignas(16) unsigned short As[2][128 * 32];
  __shared__ alignas(16) unsigned short Bs[2][64 * 32];
  A    += (size_t)blockIdx.z * sA;
  Bsrc += (size_t)blockIdx.z * sB;
  if (Cf)    Cf    += (size_t)blockIdx.z * sC;
  if (Cb)    Cb    += (size_t)blockIdx.z * sC;
  if (resid) resid += (size_t)blockIdx.z * sC;

  const int m0 = blockIdx.y * 128, n0 = blockIdx.x * 64;
  const int tid = threadIdx.x, lane = tid & 31, wave = tid >> 5;
  const int wr = wave >> 1, wc = wave & 1;        // 4x2 wave grid, 32x32 each
  const int half = lane >> 4, r = lane & 15;
  const int nrows = (Nr - n0 < 64) ? (Nr - n0) : 64;
  const unsigned as_off = (unsigned)(size_t)(void*)&As[0][0];
  const unsigned bs_off = (unsigned)(size_t)(void*)&Bs[0][0];

  v8f acc[2][2] = {};

  if (transB) {
    // ---- double-buffered Tensor-Data-Mover pipeline ----
    if (wave == 0) {
      tdm_load_2d(as_off, A    + (size_t)m0 * Kr, 32u, 128u, 128u, (unsigned)Kr);
      tdm_load_2d(bs_off, Bsrc + (size_t)n0 * Kr, 32u, 64u,
                  (unsigned)nrows, (unsigned)Kr);
      __builtin_amdgcn_s_wait_tensorcnt(0);
    }
    __syncthreads();
    const int nsteps = Kr >> 5;
    for (int s = 0; s < nsteps; ++s) {
      const int cur = s & 1;
      if (wave == 0 && s + 1 < nsteps) {        // prefetch next tile (async)
        const int k1 = (s + 1) << 5;
        tdm_load_2d(as_off + (unsigned)(cur ^ 1) * (128u * 32u * 2u),
                    A + (size_t)m0 * Kr + k1, 32u, 128u, 128u, (unsigned)Kr);
        tdm_load_2d(bs_off + (unsigned)(cur ^ 1) * (64u * 32u * 2u),
                    Bsrc + (size_t)n0 * Kr + k1, 32u, 64u,
                    (unsigned)nrows, (unsigned)Kr);
      }
      frag_mma(&As[cur][0], &Bs[cur][0], wr, wc, half, r, acc);
      if (wave == 0) __builtin_amdgcn_s_wait_tensorcnt(0);
      __syncthreads();
    }
  } else {
    // ---- manual transposing staging (batched NN mixes) ----
    for (int k0 = 0; k0 < Kr; k0 += 32) {
      for (int i = tid; i < 128 * 32; i += 256) {
        const int mm = i >> 5, kk = i & 31;
        As[0][i] = A[(size_t)(m0 + mm) * Kr + k0 + kk];
      }
      for (int i = tid; i < 64 * 32; i += 256) {
        const int nn = i >> 5, kk = i & 31;
        const int gn = n0 + nn;
        Bs[0][i] = (gn < Nr) ? Bsrc[(size_t)(k0 + kk) * Nr + gn]
                             : (unsigned short)0;
      }
      __syncthreads();
      frag_mma(&As[0][0], &Bs[0][0], wr, wc, half, r, acc);
      __syncthreads();
    }
  }

#pragma unroll
  for (int mi = 0; mi < 2; ++mi)
#pragma unroll
  for (int ci = 0; ci < 2; ++ci) {
    const int gcol = n0 + wc * 32 + ci * 16 + r;
    if (gcol >= Nr) continue;
#pragma unroll
    for (int j = 0; j < 8; ++j) {
      const int grow = m0 + wr * 32 + mi * 16 + j + 8 * half;
      float v = acc[mi][ci][j];
      if (bias) v += bias[gcol];
      if (act == 1)      v = 0.5f * v * (1.0f + erff(v * 0.70710678118f));
      else if (act == 2) v = fminf(40.0f, fmaxf(-40.0f, v));
      if (resid) v += resid[(size_t)grow * ldc + gcol];
      if (Cf) Cf[(size_t)grow * ldc + gcol] = v;
      if (Cb) Cb[(size_t)grow * ldc + gcol] = f2bf(v);
    }
  }
}

// ---------------------------------------------------------------------------
// fp32 -> bf16 bulk conversion (weights / inputs), one element per thread.
// ---------------------------------------------------------------------------
__global__ void cvt_k(const float* __restrict__ in, unsigned short* __restrict__ out,
                      long long n)
{
  const long long i = (long long)blockIdx.x * 256 + threadIdx.x;
  if (i < n) out[i] = f2bf(in[i]);
}

// ---------------------------------------------------------------------------
// LayerNorm (weight only, eps=1e-5), one row per block; fp32 and/or bf16 out.
// ---------------------------------------------------------------------------
__global__ __launch_bounds__(256)
void ln_k(const float* __restrict__ x, const float* __restrict__ w,
          float* __restrict__ of, unsigned short* __restrict__ ob)
{
  const int row = blockIdx.x;
  const float* xr = x + (size_t)row * Hdim;
  __shared__ float red[256];
  float s = 0.0f;
  for (int i = threadIdx.x; i < Hdim; i += 256) s += xr[i];
  red[threadIdx.x] = s; __syncthreads();
  for (int off = 128; off > 0; off >>= 1) {
    if (threadIdx.x < off) red[threadIdx.x] += red[threadIdx.x + off];
    __syncthreads();
  }
  const float mean = red[0] * (1.0f / Hdim); __syncthreads();
  float v = 0.0f;
  for (int i = threadIdx.x; i < Hdim; i += 256) {
    float d = xr[i] - mean; v += d * d;
  }
  red[threadIdx.x] = v; __syncthreads();
  for (int off = 128; off > 0; off >>= 1) {
    if (threadIdx.x < off) red[threadIdx.x] += red[threadIdx.x + off];
    __syncthreads();
  }
  const float inv = rsqrtf(red[0] * (1.0f / Hdim) + 1e-5f); __syncthreads();
  for (int i = threadIdx.x; i < Hdim; i += 256) {
    const float val = (xr[i] - mean) * inv * w[i];
    if (of) of[(size_t)row * Hdim + i] = val;
    if (ob) ob[(size_t)row * Hdim + i] = f2bf(val);
  }
}

// ---------------------------------------------------------------------------
// Fused causal attention on packed bf16 qkv [B*T, 3H].
// One wave per (16-query tile, head, batch); S in LDS fp32, P re-quantized.
// ---------------------------------------------------------------------------
__global__ __launch_bounds__(32)
void attn_fused(const unsigned short* __restrict__ qkv,
                unsigned short* __restrict__ out)
{
  const int qt = blockIdx.x, h = blockIdx.y, b = blockIdx.z;
  const int lane = threadIdx.x, half = lane >> 4, r = lane & 15;
  const int H3 = 3 * Hdim;
  const size_t base = (size_t)b * Tdim * H3;
  const int qoff = h * HDdim, koff = Hdim + h * HDdim, voff = 2 * Hdim + h * HDdim;

  __shared__ float          S[16][Tdim];
  __shared__ unsigned short P[16][Tdim];

  U16 aq[2];
  const int qrow = qt * 16 + r;
  for (int ks = 0; ks < 2; ++ks) {
    const unsigned short* qp = qkv + base + (size_t)qrow * H3 + qoff + ks * 32;
#pragma unroll
    for (int i = 0; i < 16; ++i) aq[ks].u[i] = qp[half * 16 + i];
  }
  for (int nt = 0; nt < Tdim / 16; ++nt) {
    v8f acc = {};
    const int kcol = nt * 16 + r;
    const unsigned short* kp = qkv + base + (size_t)kcol * H3 + koff;
    for (int ks = 0; ks < 2; ++ks) {
      U16 bk;
#pragma unroll
      for (int i = 0; i < 16; ++i) bk.u[i] = kp[ks * 32 + half * 16 + i];
      acc = wmma_bf(aq[ks], bk, acc);
    }
#pragma unroll
    for (int j = 0; j < 8; ++j) {
      const int qi = qt * 16 + j + 8 * half;
      const int kj = nt * 16 + r;
      float v = acc[j] * SCALE_;
      if (kj > qi) v = NEG_;
      S[j + 8 * half][kj] = v;
    }
  }
  __syncthreads();
  if (half == 0) {                        // lanes 0..15: one softmax row each
    float mx = -1e30f;
    for (int j = 0; j < Tdim; ++j) mx = fmaxf(mx, S[r][j]);
    float sum = 0.0f;
    for (int j = 0; j < Tdim; ++j) { float e = __expf(S[r][j] - mx); S[r][j] = e; sum += e; }
    const float inv = 1.0f / sum;
    for (int j = 0; j < Tdim; ++j) P[r][j] = f2bf(S[r][j] * inv);
  }
  __syncthreads();
  for (int ot = 0; ot < HDdim / 16; ++ot) {
    v8f acc = {};
    for (int kt = 0; kt < Tdim / 32; ++kt) {
      U16 ap, bv;
#pragma unroll
      for (int i = 0; i < 16; ++i) ap.u[i] = P[r][kt * 32 + half * 16 + i];
#pragma unroll
      for (int i = 0; i < 16; ++i) {
        const int kk = kt * 32 + half * 16 + i;
        bv.u[i] = qkv[base + (size_t)kk * H3 + voff + ot * 16 + r];
      }
      acc = wmma_bf(ap, bv, acc);
    }
#pragma unroll
    for (int j = 0; j < 8; ++j) {
      const int qi = qt * 16 + j + 8 * half;
      out[((size_t)b * Tdim + qi) * Hdim + h * HDdim + ot * 16 + r] = f2bf(acc[j]);
    }
  }
}

// ---------------------------------------------------------------------------
// masked_ctx mixing weights from bf16 qkv: per (b,t) row ->
//   yw[b,t,:] = softmax((mask(clip(meanH softmax(strict-causal s))) + g)/TAU)
// ---------------------------------------------------------------------------
__global__ __launch_bounds__(256)
void mctx_w(const unsigned short* __restrict__ qkv, unsigned short* __restrict__ yw,
            unsigned seed)
{
  const int t = blockIdx.x, b = blockIdx.y, s = threadIdx.x;
  const int H3 = 3 * Hdim;
  const size_t base = (size_t)b * Tdim * H3;
  __shared__ float qsh[Hdim];
  __shared__ float red[256];
  for (int i = s; i < Hdim; i += 256) qsh[i] = bf2f(qkv[base + (size_t)t * H3 + i]);
  __syncthreads();

  float sc[NHdim];
  for (int h = 0; h < NHdim; ++h) {
    const unsigned short* kp = qkv + base + (size_t)s * H3 + Hdim + h * HDdim;
    float d = 0.0f;
    for (int e = 0; e < HDdim; ++e) d += qsh[h * HDdim + e] * bf2f(kp[e]);
    sc[h] = (s < t) ? d * SCALE_ : NEG_;
  }
  float wmean = 0.0f;
  for (int h = 0; h < NHdim; ++h) {
    red[s] = sc[h]; __syncthreads();
    for (int off = 128; off > 0; off >>= 1) { if (s < off) red[s] = fmaxf(red[s], red[s + off]); __syncthreads(); }
    const float mx = red[0]; __syncthreads();
    const float e = __expf(sc[h] - mx);
    red[s] = e; __syncthreads();
    for (int off = 128; off > 0; off >>= 1) { if (s < off) red[s] += red[s + off]; __syncthreads(); }
    const float sum = red[0]; __syncthreads();
    wmean += e / sum;
  }
  wmean *= (1.0f / NHdim);
  wmean = fminf(40.0f, fmaxf(-40.0f, wmean));
  float lg = (((s < t) ? wmean : NEG_) +
              gumbelf(seed, (unsigned)(((b * Tdim) + t) * Tdim + s))) * 0.5f;
  red[s] = lg; __syncthreads();
  for (int off = 128; off > 0; off >>= 1) { if (s < off) red[s] = fmaxf(red[s], red[s + off]); __syncthreads(); }
  const float mx = red[0]; __syncthreads();
  const float e = __expf(lg - mx);
  red[s] = e; __syncthreads();
  for (int off = 128; off > 0; off >>= 1) { if (s < off) red[s] += red[s + off]; __syncthreads(); }
  yw[((size_t)b * Tdim + t) * Tdim + s] = f2bf(e / red[0]);
}

// ------------------------- small helper kernels ----------------------------
__global__ void zero_t0_bf(unsigned short* o) {
  const int idx = blockIdx.x * 256 + threadIdx.x;
  if (idx < Bdim * Hdim) {
    const int b = idx / Hdim, h = idx % Hdim;
    o[((size_t)b * Tdim) * Hdim + h] = 0;
  }
}
__global__ void init_ne(const float* __restrict__ ne0, float* __restrict__ nef,
                        unsigned short* __restrict__ neb) {
  const size_t idx = (size_t)blockIdx.x * 256 + threadIdx.x;
  if (idx < (size_t)Mdim * 5 * Hdim) {
    const size_t m = idx / (5 * Hdim);
    const int h = (int)(idx % Hdim);
    const float v = ne0[m * Hdim + h];
    nef[idx] = v;
    neb[idx] = f2bf(v);
  }
}
__global__ void init_nv(const float* __restrict__ v0, float* __restrict__ nv) {
  const int idx = blockIdx.x * 256 + threadIdx.x;
  if (idx < Mdim * 5) nv[idx] = v0[idx / 5];
}
__global__ void pad_w(const float* __restrict__ w, unsigned short* __restrict__ wp) {
  const size_t idx = (size_t)blockIdx.x * 256 + threadIdx.x;
  if (idx < (size_t)Hdim * KPAD) {
    const size_t n = idx / KPAD; const int k = (int)(idx % KPAD);
    wp[idx] = (k < KRAW) ? f2bf(w[n * KRAW + k]) : (unsigned short)0;
  }
}
__global__ void gather_last(const float* __restrict__ ne, float* __restrict__ fb) {
  const size_t idx = (size_t)blockIdx.x * 256 + threadIdx.x;
  if (idx < (size_t)Mdim * Hdim) {
    const size_t m = idx / Hdim; const int h = (int)(idx % Hdim);
    fb[idx] = ne[(m * 5 + 4) * Hdim + h];
  }
}
// s[m,n] = q[m] . kmat[m,n] / sqrt(H); one wave per (m,n), 5 waves per block
__global__ __launch_bounds__(160)
void operand_scores(const float* __restrict__ q, const float* __restrict__ kmat,
                    float* __restrict__ sout)
{
  const int m = blockIdx.x;
  const int wv = threadIdx.x >> 5, lane = threadIdx.x & 31;
  const float* qr = q + (size_t)m * Hdim;
  const float* kr = kmat + ((size_t)m * 5 + wv) * Hdim;
  float d = 0.0f;
  for (int i = lane; i < Hdim; i += 32) d += qr[i] * kr[i];
  for (int off = 16; off > 0; off >>= 1) d += __shfl_down(d, off, 32);
  if (lane == 0) sout[m * 5 + wv] = d * 0.03608439182f;   // 1/sqrt(768)
}
__global__ void operand_soft(const float* __restrict__ sin, float* __restrict__ aout,
                             unsigned seed)
{
  const int m = blockIdx.x * blockDim.x + threadIdx.x;
  if (m >= Mdim) return;
  float v[5];
  float mx = -1e30f;
  for (int n = 0; n < 5; ++n) { v[n] = sin[m * 5 + n]; mx = fmaxf(mx, v[n]); }
  float sum = 0.0f;
  for (int n = 0; n < 5; ++n) { v[n] = __expf(v[n] - mx); sum += v[n]; }
  for (int n = 0; n < 5; ++n) {
    float a = fminf(40.0f, fmaxf(-40.0f, v[n] / sum));
    v[n] = (a + gumbelf(seed, (unsigned)(m * 5 + n))) * 0.5f;   // /TAU
  }
  mx = -1e30f;
  for (int n = 0; n < 5; ++n) mx = fmaxf(mx, v[n]);
  sum = 0.0f;
  for (int n = 0; n < 5; ++n) { v[n] = __expf(v[n] - mx); sum += v[n]; }
  for (int n = 0; n < 5; ++n) aout[m * 5 + n] = v[n] / sum + 1e-10f;
}
__global__ void ow_k(const float* __restrict__ opl, float* __restrict__ ow, unsigned seed)
{
  const int m = blockIdx.x * blockDim.x + threadIdx.x;
  if (m >= Mdim) return;
  float mx = -1e30f; int am = 0;
  for (int j = 0; j < NOPSd; ++j) {
    const float g = (opl[m * NOPSd + j] + gumbelf(seed, (unsigned)(m * NOPSd + j))) * 0.5f;
    if (g > mx) { mx = g; am = j; }
  }
  for (int j = 0; j < NOPSd; ++j)
    ow[m * NOPSd + j] = ((j == am) ? 1.0f : 0.0f) + 1e-10f;
}
__global__ void dag_combine(const float* __restrict__ a1, const float* __restrict__ a2,
                            float* __restrict__ nv, const float* __restrict__ ow,
                            float* __restrict__ nval, int step)
{
  const int m = blockIdx.x * blockDim.x + threadIdx.x;
  if (m >= Mdim) return;
  float v1 = 0.0f, v2 = 0.0f;
  for (int n = 0; n < 5; ++n) {
    v1 += a1[m * 5 + n] * nv[m * 5 + n];
    v2 += a2[m * 5 + n] * nv[m * 5 + n];
  }
  const float res[4] = { v1 + v2, v1, v1 * v2, v1 - v2 };
  float nvl = 0.0f;
  for (int j = 0; j < NOPSd; ++j) nvl += ow[m * NOPSd + j] * res[j];
  nval[m] = nvl;
  nv[m * 5 + step + 1] = nvl;
}
__global__ __launch_bounds__(256)
void feat_build(const float* __restrict__ a1, const float* __restrict__ a2,
                const float* __restrict__ ne, const float* __restrict__ nval,
                const float* __restrict__ s2e_w, const float* __restrict__ s2e_b,
                const float* __restrict__ step_emb, const float* __restrict__ ow,
                unsigned short* __restrict__ featp, int step)
{
  const int m = blockIdx.x;
  const size_t base = (size_t)m * KPAD;
  for (int h = threadIdx.x; h < Hdim; h += 256) {
    float e1 = 0.0f, e2 = 0.0f;
    for (int n = 0; n < 5; ++n) {
      const float nev = ne[((size_t)m * 5 + n) * Hdim + h];
      e1 += a1[m * 5 + n] * nev;
      e2 += a2[m * 5 + n] * nev;
    }
    featp[base + h]            = f2bf(e1);
    featp[base + Hdim + h]     = f2bf(e2);
    featp[base + 2 * Hdim + h] = f2bf(nval[m] * s2e_w[h] + s2e_b[h]);
    featp[base + 3 * Hdim + h] = f2bf(step_emb[(size_t)step * Hdim + h]);
    if (h < NOPSd)            featp[base + 4 * Hdim + h] = f2bf(ow[m * NOPSd + h]);
    if (h >= NOPSd && h < 32) featp[base + 4 * Hdim + h] = 0;   // pad 3076..3103
  }
}

// ---------------------------------------------------------------------------
extern "C" void kernel_launch(void* const* d_in, const int* in_sizes, int n_in,
                              void* d_out, int out_size, void* d_ws, size_t ws_size,
                              hipStream_t stream)
{
  (void)in_sizes; (void)n_in; (void)out_size; (void)ws_size;
  const float* x          = (const float*)d_in[0];
  const float* neb_ln1_w  = (const float*)d_in[1];
  const float* neb_attn_w = (const float*)d_in[2];
  const float* neb_proj_w = (const float*)d_in[3];
  const float* neb_ln2_w  = (const float*)d_in[4];
  const float* neb_fc_w   = (const float*)d_in[5];
  const float* neb_fc2_w  = (const float*)d_in[6];
  const float* pdb_ln1_w  = (const float*)d_in[7];
  const float* pdb_attn_w = (const float*)d_in[8];
  const float* pdb_proj_w = (const float*)d_in[9];
  const float* pdb_ln2_w  = (const float*)d_in[10];
  const float* pdb_fc_w   = (const float*)d_in[11];
  const float* pdb_fc2_w  = (const float*)d_in[12];
  const float* ve_in_w    = (const float*)d_in[13];
  const float* ve_in_b    = (const float*)d_in[14];
  const float* ve_out_w   = (const float*)d_in[15];
  const float* ve_out_b   = (const float*)d_in[16];
  const float* ve_proj_w  = (const float*)d_in[17];
  const float* ve_proj_b  = (const float*)d_in[18];
  const float* ocs_in_w   = (const float*)d_in[19];
  const float* ocs_in_b   = (const float*)d_in[20];
  const float* pcs_in_w   = (const float*)d_in[21];
  const float* pcs_in_b   = (const float*)d_in[22];
  const float* c_a1_in_w  = (const float*)d_in[23];
  const float* c_a1_in_b  = (const float*)d_in[24];
  const float* c_a2_in_w  = (const float*)d_in[25];
  const float* c_a2_in_b  = (const float*)d_in[26];
  const float* c_opq_w    = (const float*)d_in[27];
  const float* c_opq_b    = (const float*)d_in[28];
  const float* c_ops_w    = (const float*)d_in[29];
  const float* c_ops_b    = (const float*)d_in[30];
  const float* step_emb   = (const float*)d_in[31];
  const float* emlp_w     = (const float*)d_in[32];
  const float* emlp_b     = (const float*)d_in[33];
  const float* dn_w       = (const float*)d_in[34];
  const float* s2e_w      = (const float*)d_in[35];
  const float* s2e_b      = (const float*)d_in[36];

  const size_t MH  = (size_t)Mdim * Hdim;
  const size_t HH  = (size_t)Hdim * Hdim;

  // ---- fp32 scratch ----
  float* ws = (float*)d_ws;
  size_t off = 0;
  auto alloc = [&](size_t n) { float* p = ws + off; off += n; return p; };
  float* x1     = alloc(MH);
  float* ne0    = alloc(MH);
  float* v0     = alloc(Mdim);
  float* q1     = alloc(MH);
  float* q2     = alloc(MH);
  float* opl    = alloc((size_t)Mdim * NOPSd);
  float* nef    = alloc(MH * 5);
  float* nv     = alloc((size_t)Mdim * 5);
  float* kmat   = alloc(MH * 5);
  float* sdag   = alloc((size_t)Mdim * 5);
  float* a1     = alloc((size_t)Mdim * 5);
  float* a2     = alloc((size_t)Mdim * 5);
  float* ow     = alloc((size_t)Mdim * NOPSd);
  float* nval   = alloc(Mdim);
  float* finalb = alloc(MH);
  float* hpd    = alloc(MH);

  // ---- bf16 scratch (16-byte aligned chunks for TDM) ----
  unsigned short* wsb = (unsigned short*)(ws + off);
  size_t offb = 0;
  auto allocb = [&](size_t n) {
    unsigned short* p = wsb + offb; offb += (n + 15) & ~(size_t)15; return p;
  };
  unsigned short* xb     = allocb(MH);
  unsigned short* tlnb   = allocb(MH);
  unsigned short* qkvb   = allocb(MH * 3);
  unsigned short* ybufb  = allocb(MH);
  unsigned short* fmidb  = allocb(MH * 4);
  unsigned short* ne0b   = allocb(MH);
  unsigned short* vy2b   = allocb(MH);
  unsigned short* octxb  = allocb(MH);
  unsigned short* pctxb  = allocb(MH);
  unsigned short* ywb    = allocb((size_t)Bdim * Tdim * Tdim);
  unsigned short* topqb  = allocb(MH);
  unsigned short* nebf   = allocb(MH * 5);
  unsigned short* featb  = allocb((size_t)Mdim * KPAD);
  unsigned short* wpadb  = allocb((size_t)Hdim * KPAD);
  // weights
  unsigned short* w_nat  = allocb(3 * HH);
  unsigned short* w_npr  = allocb(HH);
  unsigned short* w_nfc  = allocb(4 * HH);
  unsigned short* w_nf2  = allocb(4 * HH);
  unsigned short* w_pat  = allocb(3 * HH);
  unsigned short* w_ppr  = allocb(HH);
  unsigned short* w_pfc  = allocb(4 * HH);
  unsigned short* w_pf2  = allocb(4 * HH);
  unsigned short* w_vin  = allocb(3 * HH);
  unsigned short* w_vout = allocb(HH);
  unsigned short* w_vpr  = allocb(Hdim);
  unsigned short* w_ocs  = allocb(3 * HH);
  unsigned short* w_pcs  = allocb(3 * HH);
  unsigned short* w_a1   = allocb(3 * HH);
  unsigned short* w_a2   = allocb(3 * HH);
  unsigned short* w_opq  = allocb(HH);
  unsigned short* w_ops  = allocb((size_t)NOPSd * Hdim);

  auto cvt = [&](const float* in, unsigned short* out, long long n) {
    cvt_k<<<(int)((n + 255) / 256), 256, 0, stream>>>(in, out, n);
  };
  auto gemm = [&](const unsigned short* A, const unsigned short* Bm,
                  const float* bias, const float* res, float* Cf, unsigned short* Cb,
                  int Mr, int Nr, int Kr, int act, int transB, int ldc,
                  int batch, long long stA, long long stB, long long stC) {
    dim3 grid((Nr + 63) / 64, Mr / 128, batch);
    gemm_k<<<grid, 256, 0, stream>>>(A, Bm, bias, res, Cf, Cb, Mr, Nr, Kr, act,
                                     transB, ldc, stA, stB, stC);
  };

  // 0. one-time bf16 conversions (weights + input)
  cvt(x, xb, (long long)MH);
  cvt(neb_attn_w, w_nat, 3 * HH);  cvt(neb_proj_w, w_npr, HH);
  cvt(neb_fc_w,   w_nfc, 4 * HH);  cvt(neb_fc2_w,  w_nf2, 4 * HH);
  cvt(pdb_attn_w, w_pat, 3 * HH);  cvt(pdb_proj_w, w_ppr, HH);
  cvt(pdb_fc_w,   w_pfc, 4 * HH);  cvt(pdb_fc2_w,  w_pf2, 4 * HH);
  cvt(ve_in_w,  w_vin, 3 * HH);    cvt(ve_out_w, w_vout, HH);
  cvt(ve_proj_w, w_vpr, Hdim);
  cvt(ocs_in_w, w_ocs, 3 * HH);    cvt(pcs_in_w, w_pcs, 3 * HH);
  cvt(c_a1_in_w, w_a1, 3 * HH);    cvt(c_a2_in_w, w_a2, 3 * HH);
  cvt(c_opq_w, w_opq, HH);         cvt(c_ops_w, w_ops, (long long)NOPSd * Hdim);
  pad_w<<<(int)(((size_t)Hdim * KPAD + 255) / 256), 256, 0, stream>>>(emlp_w, wpadb);

  auto block = [&](const float* xin, const float* ln1w, const unsigned short* attw,
                   const unsigned short* projw, const float* ln2w,
                   const unsigned short* fcw, const unsigned short* fc2w,
                   float* xoutf, unsigned short* xoutb) {
    ln_k<<<Mdim, 256, 0, stream>>>(xin, ln1w, nullptr, tlnb);
    gemm(tlnb, attw, nullptr, nullptr, nullptr, qkvb, Mdim, 3 * Hdim, Hdim, 0, 1, 3 * Hdim, 1, 0, 0, 0);
    attn_fused<<<dim3(Tdim / 16, NHdim, Bdim), 32, 0, stream>>>(qkvb, ybufb);
    gemm(ybufb, projw, nullptr, xin, x1, nullptr, Mdim, Hdim, Hdim, 0, 1, Hdim, 1, 0, 0, 0);
    ln_k<<<Mdim, 256, 0, stream>>>(x1, ln2w, nullptr, tlnb);
    gemm(tlnb, fcw, nullptr, nullptr, nullptr, fmidb, Mdim, 4 * Hdim, Hdim, 1, 1, 4 * Hdim, 1, 0, 0, 0);
    gemm(fmidb, fc2w, nullptr, x1, xoutf, xoutb, Mdim, Hdim, 4 * Hdim, 0, 1, Hdim, 1, 0, 0, 0);
  };
  auto mctx = [&](const unsigned short* Win, const float* bin, unsigned seed,
                  unsigned short* outb) {
    gemm(xb, Win, bin, nullptr, nullptr, qkvb, Mdim, 3 * Hdim, Hdim, 0, 1, 3 * Hdim, 1, 0, 0, 0);
    mctx_w<<<dim3(Tdim, Bdim), 256, 0, stream>>>(qkvb, ywb, seed);
    gemm(ywb, xb, nullptr, nullptr, nullptr, outb, Tdim, Hdim, Tdim, 0, 0, Hdim,
         Bdim, (long long)Tdim * Tdim, (long long)Tdim * Hdim, (long long)Tdim * Hdim);
    zero_t0_bf<<<(Bdim * Hdim + 255) / 256, 256, 0, stream>>>(outb);
  };

  // 1. neural-embedding block
  block(x, neb_ln1_w, w_nat, w_npr, neb_ln2_w, w_nfc, w_nf2, ne0, ne0b);

  // 2. value extractor -> v0
  gemm(ne0b, w_vin, ve_in_b, nullptr, nullptr, qkvb, Mdim, 3 * Hdim, Hdim, 0, 1, 3 * Hdim, 1, 0, 0, 0);
  attn_fused<<<dim3(Tdim / 16, NHdim, Bdim), 32, 0, stream>>>(qkvb, ybufb);
  gemm(ybufb, w_vout, ve_out_b, nullptr, nullptr, vy2b, Mdim, Hdim, Hdim, 0, 1, Hdim, 1, 0, 0, 0);
  gemm(vy2b, w_vpr, ve_proj_b, nullptr, v0, nullptr, Mdim, 1, Hdim, 0, 1, 1, 1, 0, 0, 0);

  // 3. operand / op-selection contexts
  mctx(w_ocs, ocs_in_b, 0xA001u, octxb);
  mctx(w_pcs, pcs_in_b, 0xA002u, pctxb);

  // 4. op logits (clamped) and loop-invariant queries
  gemm(pctxb, w_opq, c_opq_b, nullptr, nullptr, topqb, Mdim, Hdim, Hdim, 0, 1, Hdim, 1, 0, 0, 0);
  gemm(topqb, w_ops, c_ops_b, nullptr, opl, nullptr, Mdim, NOPSd, Hdim, 2, 1, NOPSd, 1, 0, 0, 0);
  gemm(octxb, w_a1, c_a1_in_b, nullptr, q1, nullptr, Mdim, Hdim, Hdim, 0, 1, Hdim, 1, 0, 0, 0);
  gemm(octxb, w_a2, c_a2_in_b, nullptr, q2, nullptr, Mdim, Hdim, Hdim, 0, 1, Hdim, 1, 0, 0, 0);

  // 5. DAG state init
  init_ne<<<(int)((MH * 5 + 255) / 256), 256, 0, stream>>>(ne0, nef, nebf);
  init_nv<<<(Mdim * 5 + 255) / 256, 256, 0, stream>>>(v0, nv);

  // 6. DAG steps
  for (int step = 0; step < Ddim; ++step) {
    const unsigned sd = 0xB000u + (unsigned)step * 0x101u;
    gemm(nebf, w_a1 + HH, c_a1_in_b + Hdim, nullptr, kmat, nullptr,
         Mdim * 5, Hdim, Hdim, 0, 1, Hdim, 1, 0, 0, 0);
    operand_scores<<<Mdim, 160, 0, stream>>>(q1, kmat, sdag);
    operand_soft<<<(Mdim + 255) / 256, 256, 0, stream>>>(sdag, a1, sd);
    gemm(nebf, w_a2 + HH, c_a2_in_b + Hdim, nullptr, kmat, nullptr,
         Mdim * 5, Hdim, Hdim, 0, 1, Hdim, 1, 0, 0, 0);
    operand_scores<<<Mdim, 160, 0, stream>>>(q2, kmat, sdag);
    operand_soft<<<(Mdim + 255) / 256, 256, 0, stream>>>(sdag, a2, sd + 1);
    ow_k<<<(Mdim + 255) / 256, 256, 0, stream>>>(opl, ow, sd + 2);
    dag_combine<<<(Mdim + 255) / 256, 256, 0, stream>>>(a1, a2, nv, ow, nval, step);
    feat_build<<<Mdim, 256, 0, stream>>>(a1, a2, nef, nval, s2e_w, s2e_b,
                                         step_emb, ow, featb, step);
    gemm(featb, wpadb, emlp_b, nullptr, nef + (size_t)(step + 1) * Hdim,
         nebf + (size_t)(step + 1) * Hdim, Mdim, Hdim, KPAD, 0, 1, 5 * Hdim, 1, 0, 0, 0);
  }

  // 7. final column -> prediction block -> output layernorm
  gather_last<<<(int)((MH + 255) / 256), 256, 0, stream>>>(nef, finalb);
  block(finalb, pdb_ln1_w, w_pat, w_ppr, pdb_ln2_w, w_pfc, w_pf2, hpd, nullptr);
  ln_k<<<Mdim, 256, 0, stream>>>(hpd, dn_w, (float*)d_out, nullptr);
}
